// MultiHeadAttention_64424509440104
// MI455X (gfx1250) — matrix operand order repaired and checked by reading, not compiled, not run on hardware
//
#include <hip/hip_runtime.h>
#include <hip/hip_bf16.h>

typedef __bf16 bf16_t;
typedef __attribute__((ext_vector_type(16))) __bf16 v16bf;
typedef __attribute__((ext_vector_type(8)))  __bf16 v8bf;
typedef __attribute__((ext_vector_type(8)))  float   v8f;
typedef __attribute__((ext_vector_type(4)))  unsigned int u32x4;
typedef __attribute__((ext_vector_type(8)))  int i32x8;
typedef __attribute__((ext_vector_type(4)))  int i32x4;

#define D_MODEL  1024
#define N_HEAD   16
#define HEAD_DIM 64
#define BATCH    2
#define SEQ      2048
#define TOKENS   (BATCH * SEQ)   // 4096

#if defined(__has_builtin)
#if __has_builtin(__builtin_amdgcn_tensor_load_to_lds)
#define ATHENA_TDM 1
#endif
#endif
#ifndef ATHENA_TDM
#define ATHENA_TDM 0
#endif

// ---------- bf16 helpers (round-to-nearest-even) ----------
__device__ __forceinline__ unsigned short f2bfbits(float f) {
  union { float f; unsigned u; } a; a.f = f;
  unsigned u = a.u;
  u += 0x7fffu + ((u >> 16) & 1u);
  return (unsigned short)(u >> 16);
}
__device__ __forceinline__ unsigned pack2bf(float lo, float hi) {
  return (unsigned)f2bfbits(lo) | ((unsigned)f2bfbits(hi) << 16);
}

// ---------- WMMA fragment loaders (ISA 7.12.2 layouts, wave32) ----------
// A 16x32 bf16: lane row = lane%16; K = kh*8+{0..7} and 16+kh*8+{0..7}
__device__ __forceinline__ v16bf load_frag_a(const bf16_t* base, int ld) {
  const int lane = threadIdx.x & 31;
  const int row  = lane & 15;
  const int kh   = lane >> 4;
  const bf16_t* p = base + row * ld + kh * 8;
  union { v16bf v; v8bf h[2]; } u;
  u.h[0] = *(const v8bf*)(p);
  u.h[1] = *(const v8bf*)(p + 16);
  return u.v;
}
// B 32x16 bf16 from [col][k] storage: lane col = lane%16; K = kh*8+{0..7} and 16+kh*8+{0..7}
__device__ __forceinline__ v16bf load_frag_b(const bf16_t* base, int ld) {
  const int lane = threadIdx.x & 31;
  const int col  = lane & 15;
  const int kh   = lane >> 4;
  const bf16_t* p = base + col * ld + kh * 8;
  union { v16bf v; v8bf h[2]; } u;
  u.h[0] = *(const v8bf*)(p);
  u.h[1] = *(const v8bf*)(p + 16);
  return u.v;
}
__device__ __forceinline__ v8f wmma_bf16(v16bf a, v16bf b, v8f c) {
  return __builtin_amdgcn_wmma_f32_16x16x32_bf16(false, a, false, b, (short)0, c,
                                                 false, false);
}

// =====================================================================
// Precision-convert kernels (run once per launch; everything lives in L2)
// =====================================================================
__global__ __launch_bounds__(256) void cvt_act_kernel(
    const float* __restrict__ q, const float* __restrict__ k,
    const float* __restrict__ v, unsigned* __restrict__ qb,
    unsigned* __restrict__ kb, unsigned* __restrict__ vb) {
  const int i = blockIdx.x * 256 + threadIdx.x;  // bf16-pair index
  if (i < TOKENS * D_MODEL / 2) {
    float2 a = ((const float2*)q)[i]; qb[i] = pack2bf(a.x, a.y);
    float2 b = ((const float2*)k)[i]; kb[i] = pack2bf(b.x, b.y);
    float2 c = ((const float2*)v)[i]; vb[i] = pack2bf(c.x, c.y);
  }
}

// W [K][N] fp32 -> Wt [N][K] bf16 (pre-transposed so GEMM B-tiles are K-major)
__global__ __launch_bounds__(256) void cvt_wt_kernel(
    const float* __restrict__ W0, const float* __restrict__ W1,
    const float* __restrict__ W2, const float* __restrict__ W3,
    unsigned* __restrict__ O0, unsigned* __restrict__ O1,
    unsigned* __restrict__ O2, unsigned* __restrict__ O3) {
  const float* W; unsigned* O;
  switch (blockIdx.y) {
    case 0:  W = W0; O = O0; break;
    case 1:  W = W1; O = O1; break;
    case 2:  W = W2; O = O2; break;
    default: W = W3; O = O3; break;
  }
  const int KH = D_MODEL / 2;
  const int i  = blockIdx.x * 256 + threadIdx.x;  // = n*KH + j
  if (i < D_MODEL * KH) {
    const int n = i / KH, j = i - n * KH;
    const float a = W[(2 * j) * D_MODEL + n];
    const float b = W[(2 * j + 1) * D_MODEL + n];
    O[i] = pack2bf(a, b);
  }
}

// =====================================================================
// TDM tile load: [128 rows x 32 k] bf16 tile, row stride K elements,
// LDS padded to 40-element rows (pad 16B after every 64B stored).
// D# layout per CDNA5 ISA.
// 6-arg builtin form (clang-23 / therock-10.0 toolchain).
// =====================================================================
#if ATHENA_TDM
__device__ __forceinline__ void tdm_load_tile(unsigned lds_off,
                                              const unsigned short* gptr,
                                              int nrows_total, int K) {
  unsigned long long ga = (unsigned long long)(size_t)gptr;
  u32x4 g0;
  g0.x = 1u;                                   // count=1 (valid descriptor)
  g0.y = lds_off;                              // lds_addr
  g0.z = (unsigned)ga;                         // global_addr[31:0]
  g0.w = (unsigned)((ga >> 32) & 0x01FFFFFFull) | (2u << 30);  // addr[56:32], type=2
  i32x8 g1;
  g1[0] = (int)((1u << 16)      // data_size = 2 bytes
              | (1u << 20)      // pad_enable
              | (3u << 22)      // pad_interval: every 16 DWORDs (64B row)
              | (3u << 25));    // pad_amount: 4 DWORDs (16B) -> 80B LDS rows
  g1[1] = (int)(((unsigned)K & 0xFFFFu) << 16);                 // tensor_dim0 lo
  g1[2] = (int)(((unsigned)K >> 16) |
                (((unsigned)nrows_total & 0xFFFFu) << 16));     // dim0 hi / dim1 lo
  g1[3] = (int)((((unsigned)nrows_total >> 16) & 0xFFFFu) |
                (32u << 16));                                   // dim1 hi / tile_dim0
  g1[4] = 128;                                                  // tile_dim1 (tile_dim2=0)
  g1[5] = K;                                                    // tensor_dim0_stride lo
  g1[6] = 0;
  g1[7] = 0;
  i32x4 z4 = {0, 0, 0, 0};
  i32x8 z8 = {0, 0, 0, 0, 0, 0, 0, 0};
  __builtin_amdgcn_tensor_load_to_lds(g0, g1, z4, z4, z8, 0);
}
__device__ __forceinline__ void tdm_wait() {
#if __has_builtin(__builtin_amdgcn_s_wait_tensorcnt)
  __builtin_amdgcn_s_wait_tensorcnt(0);
#else
  asm volatile("s_wait_tensorcnt 0" ::: "memory");
#endif
}
#endif

// =====================================================================
// GEMM: C[M x N] = A[M x K] * Bt[N x K]^T + bias    (BM=BN=128, BK=32)
// A, Bt pre-converted bf16. 8 waves, each owns 32x64 -> 2x4 accum tiles.
// =====================================================================
template <bool OUT_BF16>
__global__ __launch_bounds__(256) void gemm_bf16_kernel(
    const unsigned short* __restrict__ A, const unsigned short* __restrict__ Bt,
    const float* __restrict__ bias, void* __restrict__ Cv,
    int M, int N, int K) {
  constexpr int LDT = 40;  // 32 + 8 pad: 80B rows (16B-aligned fragments)
  __shared__ __align__(16) unsigned short As[128 * LDT];
  __shared__ __align__(16) unsigned short Bs[128 * LDT];

  const int tid  = threadIdx.x;
  const int lane = tid & 31;
  const int wave = tid >> 5;
  const int wm   = wave & 3;   // 4 row groups of 32
  const int wn   = wave >> 2;  // 2 col groups of 64
  const int bx   = blockIdx.x;
  const int by   = blockIdx.y;

  v8f acc[2][4];
  const v8f vzero = {0.f, 0.f, 0.f, 0.f, 0.f, 0.f, 0.f, 0.f};
#pragma unroll
  for (int rt = 0; rt < 2; ++rt)
#pragma unroll
    for (int ct = 0; ct < 4; ++ct) acc[rt][ct] = vzero;

#if ATHENA_TDM
  const unsigned as_off = (unsigned)(size_t)(void*)&As[0];
  const unsigned bs_off = (unsigned)(size_t)(void*)&Bs[0];
#endif

  for (int k0 = 0; k0 < K; k0 += 32) {
#if ATHENA_TDM
    if (wave == 0) {  // TDM is a wave-level DMA; one wave issues both tiles
      tdm_load_tile(as_off, A  + (size_t)(by * 128) * K + k0, M, K);
      tdm_load_tile(bs_off, Bt + (size_t)(bx * 128) * K + k0, N, K);
      tdm_wait();
    }
#else
    {  // fallback: vectorized copy, 2 threads per 32-element row
      const int r = tid >> 1, c0 = (tid & 1) * 16;
      const unsigned short* pa = A + (size_t)(by * 128 + r) * K + k0 + c0;
      *(uint4*)&As[r * LDT + c0]     = *(const uint4*)pa;
      *(uint4*)&As[r * LDT + c0 + 8] = *(const uint4*)(pa + 8);
      const unsigned short* pb = Bt + (size_t)(bx * 128 + r) * K + k0 + c0;
      *(uint4*)&Bs[r * LDT + c0]     = *(const uint4*)pb;
      *(uint4*)&Bs[r * LDT + c0 + 8] = *(const uint4*)(pb + 8);
    }
#endif
    if (k0 + 32 < K)
      __builtin_prefetch(&Bt[(size_t)(bx * 128 + (tid >> 1)) * K + k0 + 32], 0, 1);
    __syncthreads();

    v16bf af[2], bfr[4];
#pragma unroll
    for (int rt = 0; rt < 2; ++rt)
      af[rt] = load_frag_a((const bf16_t*)&As[(wm * 32 + rt * 16) * LDT], LDT);
#pragma unroll
    for (int ct = 0; ct < 4; ++ct)
      bfr[ct] = load_frag_b((const bf16_t*)&Bs[(wn * 64 + ct * 16) * LDT], LDT);
#pragma unroll
    for (int rt = 0; rt < 2; ++rt)
#pragma unroll
      for (int ct = 0; ct < 4; ++ct)
        acc[rt][ct] = wmma_bf16(af[rt], bfr[ct], acc[rt][ct]);
    __syncthreads();
  }

  // store: C layout -> col = lane%16, row = i + (lane/16)*8
#pragma unroll
  for (int rt = 0; rt < 2; ++rt)
#pragma unroll
    for (int ct = 0; ct < 4; ++ct) {
      const int col = bx * 128 + wn * 64 + ct * 16 + (lane & 15);
      const float bv = bias[col];
#pragma unroll
      for (int i = 0; i < 8; ++i) {
        const int row = by * 128 + wm * 32 + rt * 16 + i + (lane >> 4) * 8;
        const float v = acc[rt][ct][i] + bv;
        if (OUT_BF16)
          ((unsigned short*)Cv)[row * N + col] = f2bfbits(v);
        else
          ((float*)Cv)[row * N + col] = v;
      }
    }
}

// =====================================================================
// Flash attention: one wave per 16-query tile per (b,h).
// =====================================================================
__global__ __launch_bounds__(256) void attn_kernel(
    const unsigned short* __restrict__ qh, const unsigned short* __restrict__ kh,
    const unsigned short* __restrict__ vh, unsigned short* __restrict__ ao) {
  constexpr float SCALE = 0.125f;  // 1/sqrt(64)
  __shared__ __align__(16) unsigned short Pb[8][16 * 40];  // per-wave P tile
  __shared__ __align__(16) unsigned short Vt[8][64 * 40];  // per-wave V^T [d][key]

  const int lane = threadIdx.x & 31;
  const int wave = threadIdx.x >> 5;
  const int gid  = blockIdx.x * 8 + wave;   // 0..4095
  const int qt   = gid & 127;               // S/16 = 128 query tiles
  const int hb   = gid >> 7;
  const int h    = hb & 15;
  const int b    = hb >> 4;
  const int tok0 = b * SEQ + qt * 16;

  v16bf qf[2];
#pragma unroll
  for (int dc = 0; dc < 2; ++dc)
    qf[dc] = load_frag_a(
        (const bf16_t*)(qh + tok0 * D_MODEL + h * HEAD_DIM + dc * 32), D_MODEL);

  float mrow[8], lrow[8];
  v8f oacc[4];
  const v8f vzero = {0.f, 0.f, 0.f, 0.f, 0.f, 0.f, 0.f, 0.f};
#pragma unroll
  for (int i = 0; i < 8; ++i) { mrow[i] = -1e30f; lrow[i] = 0.f; }
#pragma unroll
  for (int t = 0; t < 4; ++t) oacc[t] = vzero;

  for (int kb = 0; kb < SEQ / 32; ++kb) {
    const int s0 = kb * 32;

    // ---- stage V^T tile: lane = (key-group of 4, d-group of 16) -> b64 stores
    {
      const int kgrp = lane & 7;   // keys 4*kgrp .. +3
      const int dgrp = lane >> 3;  // d 16*dgrp .. +15
      unsigned short kv[4][16];
#pragma unroll
      for (int kk = 0; kk < 4; ++kk) {
        const unsigned short* vp = vh +
            (b * SEQ + s0 + kgrp * 4 + kk) * D_MODEL + h * HEAD_DIM + dgrp * 16;
        union { uint4 q[2]; unsigned short s[16]; } u;
        u.q[0] = *(const uint4*)(vp);
        u.q[1] = *(const uint4*)(vp + 8);
#pragma unroll
        for (int j = 0; j < 16; ++j) kv[kk][j] = u.s[j];
      }
#pragma unroll
      for (int j = 0; j < 16; ++j) {
        union { unsigned short s[4]; uint2 d; } w;
        w.s[0] = kv[0][j]; w.s[1] = kv[1][j];
        w.s[2] = kv[2][j]; w.s[3] = kv[3][j];
        *(uint2*)&Vt[wave][(dgrp * 16 + j) * 40 + kgrp * 4] = w.d;
      }
    }

    // ---- scores: two 16x16 tiles, each accumulated over d in 2 WMMAs ----
    v8f sc[2] = {vzero, vzero};
#pragma unroll
    for (int nt = 0; nt < 2; ++nt)
#pragma unroll
      for (int dc = 0; dc < 2; ++dc) {
        v16bf kf = load_frag_b(
            (const bf16_t*)(kh + (b * SEQ + s0 + nt * 16) * D_MODEL +
                            h * HEAD_DIM + dc * 32),
            D_MODEL);
        sc[nt] = wmma_bf16(qf[dc], kf, sc[nt]);
      }

    // ---- online softmax (row reductions within 16-lane halves) ----
#pragma unroll
    for (int i = 0; i < 8; ++i) {
      const float s0v = sc[0][i] * SCALE;
      const float s1v = sc[1][i] * SCALE;
      float v = fmaxf(s0v, s1v);
#pragma unroll
      for (int off = 1; off < 16; off <<= 1) v = fmaxf(v, __shfl_xor(v, off, 32));
      const float mnew  = fmaxf(mrow[i], v);
      const float alpha = __expf(mrow[i] - mnew);
      const float p0 = __expf(s0v - mnew);
      const float p1 = __expf(s1v - mnew);
      float ps = p0 + p1;
#pragma unroll
      for (int off = 1; off < 16; off <<= 1) ps += __shfl_xor(ps, off, 32);
      lrow[i] = lrow[i] * alpha + ps;
      mrow[i] = mnew;
#pragma unroll
      for (int t = 0; t < 4; ++t) oacc[t][i] = oacc[t][i] * alpha;
      const int prow = i + (lane >> 4) * 8;
      Pb[wave][prow * 40 + (lane & 15)]      = f2bfbits(p0);
      Pb[wave][prow * 40 + 16 + (lane & 15)] = f2bfbits(p1);
    }

    // per-wave LDS RAW fence (Pb + Vt written above, read as fragments below)
    asm volatile("s_wait_dscnt 0" ::: "memory");

    // ---- P x V: A = P(16x32), B = V^T tiles (32 keys x 16 d) ----
    v16bf pf = load_frag_a((const bf16_t*)&Pb[wave][0], 40);
#pragma unroll
    for (int dt = 0; dt < 4; ++dt) {
      v16bf vf = load_frag_b((const bf16_t*)&Vt[wave][dt * 16 * 40], 40);
      oacc[dt] = wmma_bf16(pf, vf, oacc[dt]);
    }
  }

  // ---- normalize and store attention output (bf16, [B,S,D] layout) ----
#pragma unroll
  for (int i = 0; i < 8; ++i) {
    const float inv = 1.0f / lrow[i];
    const int srow  = qt * 16 + i + (lane >> 4) * 8;
#pragma unroll
    for (int dt = 0; dt < 4; ++dt) {
      const int d = h * HEAD_DIM + dt * 16 + (lane & 15);
      ao[(b * SEQ + srow) * D_MODEL + d] = f2bfbits(oacc[dt][i] * inv);
    }
  }
}

// =====================================================================
extern "C" void kernel_launch(void* const* d_in, const int* in_sizes, int n_in,
                              void* d_out, int out_size, void* d_ws, size_t ws_size,
                              hipStream_t stream) {
  const float* q  = (const float*)d_in[0];
  const float* k  = (const float*)d_in[1];
  const float* v  = (const float*)d_in[2];
  const float* Wq = (const float*)d_in[3];
  const float* bq = (const float*)d_in[4];
  const float* Wk = (const float*)d_in[5];
  const float* bk = (const float*)d_in[6];
  const float* Wv = (const float*)d_in[7];
  const float* bv = (const float*)d_in[8];
  const float* Wo = (const float*)d_in[9];
  const float* bo = (const float*)d_in[10];

  const size_t AE = (size_t)TOKENS * D_MODEL;   // 4 Mi elements
  const size_t WE = (size_t)D_MODEL * D_MODEL;  // 1 Mi elements
  unsigned short* qb  = (unsigned short*)d_ws;
  unsigned short* kb  = qb + AE;
  unsigned short* vb  = kb + AE;
  unsigned short* WqT = vb + AE;
  unsigned short* WkT = WqT + WE;
  unsigned short* WvT = WkT + WE;
  unsigned short* WoT = WvT + WE;
  unsigned short* qh  = WoT + WE;
  unsigned short* kh  = qh + AE;
  unsigned short* vh  = kh + AE;
  unsigned short* ao  = vh + AE;

  cvt_act_kernel<<<(TOKENS * D_MODEL / 2) / 256, 256, 0, stream>>>(
      q, k, v, (unsigned*)qb, (unsigned*)kb, (unsigned*)vb);
  dim3 wgrid((D_MODEL * (D_MODEL / 2)) / 256, 4);
  cvt_wt_kernel<<<wgrid, 256, 0, stream>>>(Wq, Wk, Wv, Wo, (unsigned*)WqT,
                                           (unsigned*)WkT, (unsigned*)WvT,
                                           (unsigned*)WoT);

  dim3 ggrid(D_MODEL / 128, TOKENS / 128);  // (8, 32)
  gemm_bf16_kernel<true><<<ggrid, 256, 0, stream>>>(qb, WqT, bq, qh,
                                                    TOKENS, D_MODEL, D_MODEL);
  gemm_bf16_kernel<true><<<ggrid, 256, 0, stream>>>(kb, WkT, bk, kh,
                                                    TOKENS, D_MODEL, D_MODEL);
  gemm_bf16_kernel<true><<<ggrid, 256, 0, stream>>>(vb, WvT, bv, vh,
                                                    TOKENS, D_MODEL, D_MODEL);

  attn_kernel<<<(BATCH * N_HEAD * (SEQ / 16)) / 8, 256, 0, stream>>>(qh, kh, vh, ao);

  gemm_bf16_kernel<false><<<ggrid, 256, 0, stream>>>(ao, WoT, bo, d_out,
                                                     TOKENS, D_MODEL, D_MODEL);
}
